// RecInfer_73667279061368
// MI455X (gfx1250) — matrix operand order repaired and checked by reading, not compiled, not run on hardware
//
#include <hip/hip_runtime.h>
#include <hip/hip_bf16.h>
#include <float.h>

// ---------------------------------------------------------------------------
// Problem constants (match reference)
// ---------------------------------------------------------------------------
#define B_    256
#define L_    50
#define P_    10
#define C_    8
#define N_    1000000
#define D_    64
#define TOPK  20

// Scoring tiling
#define CHUNK        8192                       // items per block chunk
#define NCHUNK       ((N_ + CHUNK - 1) / CHUNK) // 123
#define STAGE_ITEMS  128                        // items per barrier stage
#define NSTAGE       (CHUNK / STAGE_ITEMS)      // 64
#define CAP          16                         // per-thread drain buffer
#define TOT_CAND     (NCHUNK * TOPK)            // 2460 candidates per user

// LDS layout for score_topk_kernel (byte offsets into one shared block)
// Score tile is TRANSPOSED: s_sc[item][user], stride 68 floats so that
// ds_store_b128 stays 16B-aligned and both phases are bank-conflict-free
// (bank = (4*item + user) mod 64).
#define SC_STRIDE    68
#define SC_BYTES     (STAGE_ITEMS * SC_STRIDE * 4)   // 34816
#define BUFS_OFF     SC_BYTES
#define BUFI_OFF     (SC_BYTES + 256 * CAP * 4)      // 51200
#define SMEM_BYTES   (SC_BYTES + 256 * CAP * 8)      // 67584

typedef __bf16 bf16_t;
typedef __attribute__((ext_vector_type(16))) __bf16 v16bf;
typedef __attribute__((ext_vector_type(8)))  float  v8f;
typedef __attribute__((ext_vector_type(4)))  float  v4f;

// ---------------------------------------------------------------------------
// Kernel 1: user tower.
//   p = profile @ Wp ; c = mean(context,axis=1) @ Wc ; i = mean(gathered items)
//   user = tanh((p+c+i) @ Wu + bu)
// ---------------------------------------------------------------------------
__global__ __launch_bounds__(64) void user_tower_kernel(
    const int* __restrict__ profile, const int* __restrict__ context,
    const int* __restrict__ item_idx, const float* __restrict__ items,
    const float* __restrict__ Wp, const float* __restrict__ Wc,
    const float* __restrict__ Wu, const float* __restrict__ bu,
    float* __restrict__ user_out)
{
    __shared__ float s_ctx[C_];
    __shared__ float s_prof[P_];
    __shared__ float s_h[D_];

    const int b = blockIdx.x;
    const int d = threadIdx.x;

    if (d < C_) {
        int s = 0;
        for (int l = 0; l < L_; ++l) s += context[(b * L_ + l) * C_ + d];
        s_ctx[d] = (float)s * (1.0f / (float)L_);
    }
    if (d < P_) s_prof[d] = (float)profile[b * P_ + d];
    __syncthreads();

    float p = 0.f;
#pragma unroll
    for (int k = 0; k < P_; ++k) p += s_prof[k] * Wp[k * D_ + d];

    float c = 0.f;
#pragma unroll
    for (int k = 0; k < C_; ++k) c += s_ctx[k] * Wc[k * D_ + d];

    float acc = 0.f;
    for (int l = 0; l < L_; ++l) {
        int idx = item_idx[b * L_ + l];
        acc += items[(size_t)idx * D_ + d];
    }
    acc *= (1.0f / (float)L_);

    s_h[d] = p + c + acc;
    __syncthreads();

    float u = bu[d];
#pragma unroll
    for (int k = 0; k < D_; ++k) u += s_h[k] * Wu[k * D_ + d];
    user_out[b * D_ + d] = tanhf(u);
}

// ---------------------------------------------------------------------------
// Kernel 2: dense scoring via v_wmma_f32_16x16x32_bf16 + per-chunk top-20.
// Grid (NCHUNK, 4); block 256 = 8 waves.
//   wave w: user-tile PAIR utp = w&1, item subtile is = w>>1.
// Each B fragment pair (8x global_load_b128 + cvt_pk_bf16) feeds 4 WMMAs.
// Score tile goes to LDS transposed -> 2x ds_store_b128 per 16x16 tile.
// Top-k: strict threshold filter -> per-thread LDS buffer -> rare drain().
// ---------------------------------------------------------------------------
__global__ __launch_bounds__(256) void score_topk_kernel(
    const float* __restrict__ items, const float* __restrict__ user_vec,
    float* __restrict__ cand_score, int* __restrict__ cand_idx)
{
    const int chunk = blockIdx.x;
    const int ugrp  = blockIdx.y;            // 64-user group (0..3)
    const int tid   = threadIdx.x;
    const int lane  = tid & 31;
    const int wave  = tid >> 5;
    const int utp   = wave & 1;              // user tile pair (0/1)
    const int is    = wave >> 1;             // item subtile (0..3)
    const bool hi   = lane >= 16;
    const int mrow  = lane & 15;
    const int col   = lane & 15;
    const int hioff = hi ? 8 : 0;

    __shared__ alignas(16) char smem[SMEM_BYTES];
    float* s_sc = (float*)smem;                    // [128 items][SC_STRIDE]
    float* bufS = (float*)(smem + BUFS_OFF);       // [256][CAP] drain buffer
    int*   bufI = (int*)(smem + BUFI_OFF);

    // ---- A fragments: two 16-user tiles, K = 0..63 (2 frags each) ----
    // ISA 16-bit A layout: lane<16: k = K0+j (j<8), K0+16+(j-8); lane>=16: +8
    v16bf a00, a01, a10, a11;
    {
        const int K0 = hioff;
        const float* u0 = user_vec + (size_t)(ugrp * 64 + (utp * 2 + 0) * 16 + mrow) * D_;
        const float* u1 = user_vec + (size_t)(ugrp * 64 + (utp * 2 + 1) * 16 + mrow) * D_;
#pragma unroll
        for (int j = 0; j < 8; ++j) {
            a00[j]     = (bf16_t)u0[K0 + j];
            a00[8 + j] = (bf16_t)u0[K0 + 16 + j];
            a01[j]     = (bf16_t)u0[32 + K0 + j];
            a01[8 + j] = (bf16_t)u0[32 + K0 + 16 + j];
            a10[j]     = (bf16_t)u1[K0 + j];
            a10[8 + j] = (bf16_t)u1[K0 + 16 + j];
            a11[j]     = (bf16_t)u1[32 + K0 + j];
            a11[8 + j] = (bf16_t)u1[32 + K0 + 16 + j];
        }
    }

    // ---- per-thread top-20 state (registers, static indexing only) ----
    float ts[TOPK];
    int   ti[TOPK];
#pragma unroll
    for (int i = 0; i < TOPK; ++i) { ts[i] = -FLT_MAX; ti[i] = 0x7FFFFFFF; }
    float thr = -FLT_MAX;   // == ts[TOPK-1] after each drain
    int   cnt = 0;

    auto insert = [&](float s, int id) {
        if (s > ts[TOPK - 1] || (s == ts[TOPK - 1] && id < ti[TOPK - 1])) {
            float cs = s; int ci = id;
#pragma unroll
            for (int j = 0; j < TOPK; ++j) {
                bool gt = (cs > ts[j]) || (cs == ts[j] && ci < ti[j]);
                float os = ts[j]; int oi = ti[j];
                if (gt) { ts[j] = cs; ti[j] = ci; cs = os; ci = oi; }
            }
        }
    };
    auto drain = [&]() {
        for (int c = 0; c < cnt; ++c) insert(bufS[tid * CAP + c], bufI[tid * CAP + c]);
        cnt = 0;
        thr = ts[TOPK - 1];
    };

    const int ul2 = tid & 63;   // filter phase: local user
    const int sub = tid >> 6;   // filter phase: item stripe (0..3)

    for (int st = 0; st < NSTAGE; ++st) {
        const int sb = chunk * CHUNK + st * STAGE_ITEMS;

        // ---- compute 128 items x 64 users into LDS (transposed) ----
#pragma unroll
        for (int ss = 0; ss < 2; ++ss) {
            const int item  = sb + ss * 64 + is * 16 + col;
            const int itemc = (item < N_) ? item : (N_ - 1);
            const float* bp = items + (size_t)itemc * D_ + (hi ? 8 : 0);
            __builtin_prefetch(bp + (size_t)STAGE_ITEMS * D_, 0, 1);

            // ISA 16-bit B layout: lane<16: k = K0+j (j<8), K0+16+(j-8); lane>=16: +8
            v16bf b0, b1;
#pragma unroll
            for (int j = 0; j < 16; ++j) b0[j] = (bf16_t)bp[(j < 8) ? j : (j + 8)];
#pragma unroll
            for (int j = 0; j < 16; ++j) b1[j] = (bf16_t)bp[32 + ((j < 8) ? j : (j + 8))];

            const int citem = ss * 64 + is * 16 + col;      // 0..127
            float* srow = s_sc + (size_t)citem * SC_STRIDE;
            {
                v8f c = {};
                c = __builtin_amdgcn_wmma_f32_16x16x32_bf16(
                        false, a00, false, b0, (short)0, c, false, false);
                c = __builtin_amdgcn_wmma_f32_16x16x32_bf16(
                        false, a01, false, b1, (short)0, c, false, false);
                v4f* dst = (v4f*)(srow + utp * 32 + hioff);  // users utp*32+hioff..+7
                dst[0] = (v4f){c[0], c[1], c[2], c[3]};
                dst[1] = (v4f){c[4], c[5], c[6], c[7]};
            }
            {
                v8f c = {};
                c = __builtin_amdgcn_wmma_f32_16x16x32_bf16(
                        false, a10, false, b0, (short)0, c, false, false);
                c = __builtin_amdgcn_wmma_f32_16x16x32_bf16(
                        false, a11, false, b1, (short)0, c, false, false);
                v4f* dst = (v4f*)(srow + utp * 32 + 16 + hioff);
                dst[0] = (v4f){c[0], c[1], c[2], c[3]};
                dst[1] = (v4f){c[4], c[5], c[6], c[7]};
            }
        }
        __syncthreads();

        // ---- threshold filter: 32 scores/thread, appends are rare ----
#pragma unroll 4
        for (int i = 0; i < 32; ++i) {
            const int j = sub + 4 * i;
            const int gitem = sb + j;
            const float s = s_sc[(size_t)j * SC_STRIDE + ul2];
            if ((s > thr) && (gitem < N_)) {
                bufS[tid * CAP + cnt] = s;
                bufI[tid * CAP + cnt] = gitem;
                ++cnt;
                if (cnt == CAP) drain();
            }
        }
        __syncthreads();
    }
    drain();
    __syncthreads();   // buffers + score tile dead below; alias LDS

    // ---- merge 4 stripes/user -> per-chunk top-20 candidates ----
    float* candS = (float*)smem;                   // 64*4*20*4 = 20480 B
    int*   candI = (int*)(smem + 64 * 4 * TOPK * 4);
    const int slot = ul2 * 4 + sub;
#pragma unroll
    for (int i = 0; i < TOPK; ++i) {
        candS[slot * TOPK + i] = ts[i];
        candI[slot * TOPK + i] = ti[i];
    }
    __syncthreads();

    if (tid < 64) {
        float fs[TOPK];
        int   fi[TOPK];
#pragma unroll
        for (int i = 0; i < TOPK; ++i) { fs[i] = -FLT_MAX; fi[i] = 0x7FFFFFFF; }
        const int base = tid * 4 * TOPK;
        for (int i = 0; i < 4 * TOPK; ++i) {
            float s = candS[base + i];
            int  id = candI[base + i];
            if (s > fs[TOPK - 1] || (s == fs[TOPK - 1] && id < fi[TOPK - 1])) {
                float cs = s; int ci = id;
#pragma unroll
                for (int j = 0; j < TOPK; ++j) {
                    bool gt = (cs > fs[j]) || (cs == fs[j] && ci < fi[j]);
                    float os = fs[j]; int oi = fi[j];
                    if (gt) { fs[j] = cs; fi[j] = ci; cs = os; ci = oi; }
                }
            }
        }
        const size_t uo = ((size_t)(ugrp * 64 + tid) * NCHUNK + chunk) * TOPK;
#pragma unroll
        for (int i = 0; i < TOPK; ++i) {
            cand_score[uo + i] = fs[i];
            cand_idx[uo + i]   = fi[i];
        }
    }
}

// ---------------------------------------------------------------------------
// Kernel 3: per-user final top-20 over NCHUNK*20 candidates.
// 20 rounds of parallel argmax with (score desc, index asc) tie rule.
// ---------------------------------------------------------------------------
__global__ __launch_bounds__(256) void final_topk_kernel(
    const float* __restrict__ cand_score, const int* __restrict__ cand_idx,
    int* __restrict__ out)
{
    const int u = blockIdx.x;
    __shared__ float s_s[TOT_CAND];
    __shared__ int   s_i[TOT_CAND];
    __shared__ float r_s[256];
    __shared__ int   r_i[256];

    for (int i = threadIdx.x; i < TOT_CAND; i += 256) {
        s_s[i] = cand_score[(size_t)u * TOT_CAND + i];
        s_i[i] = cand_idx[(size_t)u * TOT_CAND + i];
    }
    __syncthreads();

    for (int k = 0; k < TOPK; ++k) {
        float bs = -FLT_MAX;
        int   bi = 0x7FFFFFFF;
        for (int i = threadIdx.x; i < TOT_CAND; i += 256) {
            float s = s_s[i]; int id = s_i[i];
            if (s > bs || (s == bs && id < bi)) { bs = s; bi = id; }
        }
        r_s[threadIdx.x] = bs; r_i[threadIdx.x] = bi;
        __syncthreads();
        for (int off = 128; off > 0; off >>= 1) {
            if (threadIdx.x < off) {
                float s2 = r_s[threadIdx.x + off]; int i2 = r_i[threadIdx.x + off];
                if (s2 > r_s[threadIdx.x] ||
                    (s2 == r_s[threadIdx.x] && i2 < r_i[threadIdx.x])) {
                    r_s[threadIdx.x] = s2; r_i[threadIdx.x] = i2;
                }
            }
            __syncthreads();
        }
        if (threadIdx.x == 0) out[u * TOPK + k] = r_i[0];
        for (int i = threadIdx.x; i < TOT_CAND; i += 256) {
            if (s_i[i] == r_i[0] && s_s[i] == r_s[0]) s_s[i] = -FLT_MAX;
        }
        __syncthreads();
    }
}

// ---------------------------------------------------------------------------
// Launch
// ---------------------------------------------------------------------------
extern "C" void kernel_launch(void* const* d_in, const int* in_sizes, int n_in,
                              void* d_out, int out_size, void* d_ws, size_t ws_size,
                              hipStream_t stream) {
    (void)in_sizes; (void)n_in; (void)out_size; (void)ws_size;

    const int*   profile      = (const int*)d_in[0];
    const int*   context      = (const int*)d_in[1];
    const int*   item_indices = (const int*)d_in[2];
    const float* item_vectors = (const float*)d_in[3];
    const float* Wp           = (const float*)d_in[4];
    const float* Wc           = (const float*)d_in[5];
    const float* Wu           = (const float*)d_in[6];
    const float* bu           = (const float*)d_in[7];
    int*         out          = (int*)d_out;

    char* ws = (char*)d_ws;
    float* user_vec   = (float*)ws;                                     // 64 KB
    float* cand_score = (float*)(ws + 64 * 1024);                       // 2.52 MB
    int*   cand_idx   = (int*)(ws + 64 * 1024 +
                               sizeof(float) * (size_t)B_ * NCHUNK * TOPK);

    user_tower_kernel<<<B_, 64, 0, stream>>>(
        profile, context, item_indices, item_vectors, Wp, Wc, Wu, bu, user_vec);

    dim3 grid2(NCHUNK, 4);
    score_topk_kernel<<<grid2, 256, 0, stream>>>(
        item_vectors, user_vec, cand_score, cand_idx);

    final_topk_kernel<<<B_, 256, 0, stream>>>(cand_score, cand_idx, out);
}
